// TransformerBlock_34961033790160
// MI455X (gfx1250) — matrix operand order repaired and checked by reading, not compiled, not run on hardware
//
#include <hip/hip_runtime.h>
#include <stdint.h>

// ---------------- problem constants ----------------
static constexpr int BN   = 8;
static constexpr int LN   = 4096;
static constexpr int DN   = 1024;
static constexpr int HN   = 16;
static constexpr int HDN  = 64;
static constexpr int HIDN = 4096;
static constexpr int MROWS = BN * LN;          // 32768
static constexpr float EPSC = 1e-6f;

typedef __bf16 bf16;
typedef __attribute__((ext_vector_type(16))) __bf16 v16bf;
typedef __attribute__((ext_vector_type(8)))  __bf16 v8bf;
typedef __attribute__((ext_vector_type(8)))  float  v8f;

// ---------------- WMMA fragment loaders (CDNA5 ISA 7.12.2) ----------------
// A 16x32 bf16: lane&15 = M row; lane>>4 selects interleaved K chunks
//   elems [0..7]  = K 8*half+0..7, elems [8..15] = K 16+8*half+0..7
__device__ __forceinline__ v16bf frag_a(const bf16* sm, int ld, int row, int kofs) {
  const int lane = threadIdx.x & 31;
  const int half = lane >> 4;
  const bf16* p = sm + (row + (lane & 15)) * ld + kofs + half * 8;
  v8bf c0 = *(const v8bf*)p;
  v8bf c1 = *(const v8bf*)(p + 16);
  v16bf f;
#pragma unroll
  for (int i = 0; i < 8; ++i) { f[i] = c0[i]; f[i + 8] = c1[i]; }
  return f;
}
// B 32x16 bf16: lane&15 = N col; lane>>4 selects interleaved K chunks
//   elems [0..7]  = K 8*half+0..7, elems [8..15] = K 16+8*half+0..7
__device__ __forceinline__ v16bf frag_b(const bf16* sm, int ld, int row, int kofs) {
  const int lane = threadIdx.x & 31;
  const int half = lane >> 4;
  const bf16* p = sm + (row + (lane & 15)) * ld + kofs + half * 8;
  v8bf c0 = *(const v8bf*)p;
  v8bf c1 = *(const v8bf*)(p + 16);
  v16bf f;
#pragma unroll
  for (int i = 0; i < 8; ++i) { f[i] = c0[i]; f[i + 8] = c1[i]; }
  return f;
}

__device__ __forceinline__ float gelu_tanh(float x) {
  return 0.5f * x * (1.0f + tanhf(0.7978845608028654f * (x + 0.044715f * x * x * x)));
}

// ---------------- generic bf16 WMMA GEMM:  C = A(M,K) * Bw(N,K)^T ----------------
// Two-stage software pipeline: global->VGPR stage for tile i+1 issued before the
// WMMAs of tile i; VGPR->LDS after; ONE barrier per k-step; L2 prefetch 2 ahead.
// EPI: 0 = bias -> bf16 out, 2 = bias+GELU -> bf16,
//      3 = bias + Res -> f32 (final residual, Out may alias Res elementwise)
template <int EPI>
__global__ __launch_bounds__(256) void gemm_bf16(
    const bf16* __restrict__ A, const bf16* __restrict__ Bw,
    const float* __restrict__ bias, const float* __restrict__ Res,
    void* __restrict__ Out, int M, int N, int K) {
  __shared__ alignas(16) bf16 As[2][128 * 40];   // stride 40 bf16 = 80B (bank-friendly)
  __shared__ alignas(16) bf16 Bs[2][128 * 40];
  const int tid = threadIdx.x;
  const int m0 = blockIdx.y * 128, n0 = blockIdx.x * 128;
  const int lrow = tid >> 1, lk = (tid & 1) * 16;
  const int wid = tid >> 5, lane = tid & 31;
  const int wm = (wid >> 2) * 64, wn = (wid & 3) * 32;   // 2x4 wave grid, 64x32 per wave
  const int lofs = lrow * 40 + lk;
  const bf16* gA = A  + (size_t)(m0 + lrow) * K + lk;    // this thread's stream in A
  const bf16* gB = Bw + (size_t)(n0 + lrow) * K + lk;
  v8f acc[4][2] = {};

  // ---- prologue: fill buffer 0 with k-tile 0 ----
  {
    uint4 a0 = *(const uint4*)(gA), a1 = *(const uint4*)(gA + 8);
    uint4 b0 = *(const uint4*)(gB), b1 = *(const uint4*)(gB + 8);
    *(uint4*)&As[0][lofs]     = a0;
    *(uint4*)&As[0][lofs + 8] = a1;
    *(uint4*)&Bs[0][lofs]     = b0;
    *(uint4*)&Bs[0][lofs + 8] = b1;
  }
  __syncthreads();

  int buf = 0;
  for (int kk = 0; kk < K; kk += 32) {
    const bool last = (kk + 32 >= K);
    uint4 na0, na1, nb0, nb1;
    if (!last) {
      // stage next k-tile into VGPRs (latency hidden behind the WMMAs below)
      const bf16* pa = gA + kk + 32;
      const bf16* pb = gB + kk + 32;
      na0 = *(const uint4*)pa; na1 = *(const uint4*)(pa + 8);
      nb0 = *(const uint4*)pb; nb1 = *(const uint4*)(pb + 8);
      if (kk + 64 < K) {                    // warm L2 two tiles ahead
        __builtin_prefetch(gA + kk + 64, 0, 1);   // global_prefetch_b8
        __builtin_prefetch(gB + kk + 64, 0, 1);
      }
    }
    // ---- compute current tile from LDS ----
    v16bf af[4], bfr[2];
#pragma unroll
    for (int i = 0; i < 4; ++i) af[i] = frag_a(As[buf], 40, wm + i * 16, 0);
#pragma unroll
    for (int j = 0; j < 2; ++j) bfr[j] = frag_b(Bs[buf], 40, wn + j * 16, 0);
#pragma unroll
    for (int i = 0; i < 4; ++i)
#pragma unroll
      for (int j = 0; j < 2; ++j)
        acc[i][j] = __builtin_amdgcn_wmma_f32_16x16x32_bf16(
            false, af[i], false, bfr[j], (short)0, acc[i][j], false, false);
    // ---- commit staged tile to the other LDS buffer ----
    if (!last) {
      *(uint4*)&As[buf ^ 1][lofs]     = na0;
      *(uint4*)&As[buf ^ 1][lofs + 8] = na1;
      *(uint4*)&Bs[buf ^ 1][lofs]     = nb0;
      *(uint4*)&Bs[buf ^ 1][lofs + 8] = nb1;
    }
    __syncthreads();
    buf ^= 1;
  }

  // ---- epilogue: C VGPR layout M = v + 8*(lane>>4), N = lane&15 ----
  const int half = lane >> 4, nl = lane & 15;
#pragma unroll
  for (int i = 0; i < 4; ++i)
#pragma unroll
    for (int j = 0; j < 2; ++j)
#pragma unroll
      for (int v = 0; v < 8; ++v) {
        int row = m0 + wm + i * 16 + v + 8 * half;
        int col = n0 + wn + j * 16 + nl;
        float val = acc[i][j][v] + bias[col];
        size_t o = (size_t)row * N + col;
        if (EPI == 0)      ((bf16*)Out)[o]  = (bf16)val;
        else if (EPI == 2) ((bf16*)Out)[o]  = (bf16)gelu_tanh(val);
        else if (EPI == 3) ((float*)Out)[o] = val + Res[o];
        else               ((float*)Out)[o] = val;
      }
}

// ---------------- circular autocorrelation via WMMA + wrapped-diagonal reduce ----
// corr[bh, (t-u) mod L] += sum_d Q[t,d]*K[u,d]; block = one 128x128 tile of QK^T
__global__ __launch_bounds__(256) void autocorr_kernel(
    const bf16* __restrict__ Q, const bf16* __restrict__ Kt, float* __restrict__ corr) {
  __shared__ alignas(16) bf16 Qs[128 * 72];  // stride 72 bf16 = 144B
  __shared__ alignas(16) bf16 Ks[128 * 72];
  __shared__ float lag[256];                 // 255 used
  const int tid = threadIdx.x;
  const int bh = blockIdx.z;
  const int t0 = blockIdx.x * 128, u0 = blockIdx.y * 128;
  const bf16* Qb = Q  + (size_t)(bh >> 4) * LN * DN + (bh & 15) * HDN;
  const bf16* Kb = Kt + (size_t)(bh >> 4) * LN * DN + (bh & 15) * HDN;
  lag[tid] = 0.0f;
  const int lrow = tid >> 1, lc = (tid & 1) * 32;
  {
    const uint4* qp = (const uint4*)(Qb + (size_t)(t0 + lrow) * DN + lc);
    const uint4* kp = (const uint4*)(Kb + (size_t)(u0 + lrow) * DN + lc);
#pragma unroll
    for (int i = 0; i < 4; ++i) {
      *(uint4*)&Qs[lrow * 72 + lc + i * 8] = qp[i];
      *(uint4*)&Ks[lrow * 72 + lc + i * 8] = kp[i];
    }
  }
  __syncthreads();
  const int wid = tid >> 5, lane = tid & 31;
  const int wt = (wid >> 1) * 32, wu = (wid & 1) * 64;   // 4x2 wave grid, 32x64 per wave
  v8f acc[2][4] = {};
#pragma unroll
  for (int ks = 0; ks < 64; ks += 32) {
    v16bf af[2], bfr[4];
#pragma unroll
    for (int i = 0; i < 2; ++i) af[i] = frag_a(Qs, 72, wt + i * 16, ks);
#pragma unroll
    for (int j = 0; j < 4; ++j) bfr[j] = frag_b(Ks, 72, wu + j * 16, ks);
#pragma unroll
    for (int i = 0; i < 2; ++i)
#pragma unroll
      for (int j = 0; j < 4; ++j)
        acc[i][j] = __builtin_amdgcn_wmma_f32_16x16x32_bf16(
            false, af[i], false, bfr[j], (short)0, acc[i][j], false, false);
  }
  const int half = lane >> 4, nl = lane & 15;
#pragma unroll
  for (int i = 0; i < 2; ++i)
#pragma unroll
    for (int j = 0; j < 4; ++j)
#pragma unroll
      for (int v = 0; v < 8; ++v) {
        int trel = wt + i * 16 + v + 8 * half;
        int urel = wu + j * 16 + nl;
        atomicAdd(&lag[trel - urel + 127], acc[i][j][v]);   // ds_add_f32
      }
  __syncthreads();
  if (tid < 255) {
    int l = (t0 - u0 + tid - 127 + LN) & (LN - 1);
    atomicAdd(&corr[(size_t)bh * LN + l], lag[tid]);
  }
}

// ---------------- layernorm -> bf16 ----------------
__global__ __launch_bounds__(256) void ln_kernel(const float* __restrict__ x,
    const float* __restrict__ g, const float* __restrict__ b, bf16* __restrict__ y) {
  const int tid = threadIdx.x;
  const float* xr = x + (size_t)blockIdx.x * DN;
  __shared__ float r1[256], r2[256];
  float s = 0.f, s2 = 0.f;
#pragma unroll
  for (int k = 0; k < DN / 256; ++k) { float v = xr[tid + k * 256]; s += v; s2 += v * v; }
  r1[tid] = s; r2[tid] = s2; __syncthreads();
  for (int st = 128; st > 0; st >>= 1) {
    if (tid < st) { r1[tid] += r1[tid + st]; r2[tid] += r2[tid + st]; }
    __syncthreads();
  }
  float mean = r1[0] * (1.0f / DN);
  float var  = r2[0] * (1.0f / DN) - mean * mean;
  float rstd = rsqrtf(var + EPSC);
  bf16* yr = y + (size_t)blockIdx.x * DN;
#pragma unroll
  for (int k = 0; k < DN / 256; ++k) {
    int i = tid + k * 256;
    yr[i] = (bf16)((xr[i] - mean) * rstd * g[i] + b[i]);
  }
}

// ---------------- softmax over L with 1/HD scale (corr mean over d) ----------------
__global__ __launch_bounds__(256) void softmax_kernel(float* __restrict__ c) {
  const int tid = threadIdx.x;
  float* row = c + (size_t)blockIdx.x * LN;
  __shared__ float red[256];
  const float scale = 1.0f / (float)HDN;
  float m = -3.0e38f;
  for (int i = tid; i < LN; i += 256) m = fmaxf(m, row[i] * scale);
  red[tid] = m; __syncthreads();
  for (int s = 128; s > 0; s >>= 1) { if (tid < s) red[tid] = fmaxf(red[tid], red[tid + s]); __syncthreads(); }
  m = red[0]; __syncthreads();
  float sum = 0.f;
  for (int i = tid; i < LN; i += 256) { float e = __expf(row[i] * scale - m); row[i] = e; sum += e; }
  red[tid] = sum; __syncthreads();
  for (int s = 128; s > 0; s >>= 1) { if (tid < s) red[tid] += red[tid + s]; __syncthreads(); }
  float inv = 1.0f / red[0];
  for (int i = tid; i < LN; i += 256) row[i] *= inv;
}

// ---------------- out2d[b, d*H + h] = sum_l attn[b,h,l] * V[b,l,h,d] --------------
__global__ __launch_bounds__(256) void attnv_kernel(const float* __restrict__ attn,
    const bf16* __restrict__ V, float* __restrict__ out2d) {
  __shared__ float a_s[LN];
  __shared__ float red[256];
  const int tid = threadIdx.x;
  const int bh = blockIdx.x, b = bh >> 4, h = bh & 15;
  const float* ar = attn + (size_t)bh * LN;
  for (int i = tid; i < LN; i += 256) a_s[i] = ar[i];
  __syncthreads();
  const int d = tid & 63, g = tid >> 6;
  const bf16* Vb = V + (size_t)b * LN * DN + h * HDN + d;
  float s = 0.f;
  for (int l = g; l < LN; l += 4) s += a_s[l] * (float)Vb[(size_t)l * DN];
  red[tid] = s; __syncthreads();
  if (g == 0) {
    float t = red[d] + red[64 + d] + red[128 + d] + red[192 + d];
    out2d[b * DN + d * HN + h] = t;   // faithful "lossy" transpose-reshape
  }
}

// ---------------- proj = out2d * Wo^T + bo  (tiny: 8x1024x1024) -------------------
__global__ __launch_bounds__(256) void proj_kernel(const float* __restrict__ in,
    const float* __restrict__ Wo, const float* __restrict__ bo, float* __restrict__ proj) {
  int idx = blockIdx.x * 256 + threadIdx.x;   // 8192
  int b = idx >> 10, o = idx & 1023;
  const float* ir = in + b * DN;
  const float* wr = Wo + (size_t)o * DN;
  float s = bo[o];
  for (int i = 0; i < DN; ++i) s += ir[i] * wr[i];
  proj[idx] = s;
}

// ---------------- x2 = (x+p) - movavg25(x+p), zero-padded SAME --------------------
__global__ __launch_bounds__(256) void decomp_kernel(const float* __restrict__ x,
    const float* __restrict__ proj, float* __restrict__ x2) {
  size_t idx = (size_t)blockIdx.x * 256 + threadIdx.x;
  int d = (int)(idx & (DN - 1));
  int l = (int)((idx >> 10) & (LN - 1));
  int b = (int)(idx >> 22);
  float p = proj[b * DN + d];
  const float* xc = x + ((size_t)b * LN) * DN + d;
  float sum = 0.f; int cnt = 0;
#pragma unroll
  for (int j = -12; j <= 12; ++j) {
    int ll = l + j;
    if (ll >= 0 && ll < LN) { sum += xc[(size_t)ll * DN]; cnt++; }
  }
  float trend = (sum + (float)cnt * p) * (1.0f / 25.0f);
  x2[idx] = x[idx] + p - trend;
}

// ---------------- f32 -> bf16 weight convert ----------------
__global__ void cvt_kernel(const float* __restrict__ in, bf16* __restrict__ o, int n) {
  int i = blockIdx.x * 256 + threadIdx.x;
  if (i < n) o[i] = (bf16)in[i];
}

// ======================= host-side orchestration =======================
extern "C" void kernel_launch(void* const* d_in, const int* in_sizes, int n_in,
                              void* d_out, int out_size, void* d_ws, size_t ws_size,
                              hipStream_t stream) {
  (void)in_sizes; (void)n_in; (void)out_size; (void)ws_size;
  const float* x    = (const float*)d_in[0];
  const float* ln1w = (const float*)d_in[1];
  const float* ln1b = (const float*)d_in[2];
  const float* Wq   = (const float*)d_in[3];
  const float* bq   = (const float*)d_in[4];
  const float* Wk   = (const float*)d_in[5];
  const float* bk   = (const float*)d_in[6];
  const float* Wv   = (const float*)d_in[7];
  const float* bv   = (const float*)d_in[8];
  const float* Wo   = (const float*)d_in[9];
  const float* bo   = (const float*)d_in[10];
  const float* ln2w = (const float*)d_in[11];
  const float* ln2b = (const float*)d_in[12];
  const float* W1   = (const float*)d_in[13];
  const float* b1   = (const float*)d_in[14];
  const float* W2   = (const float*)d_in[15];
  const float* b2   = (const float*)d_in[16];
  float* out = (float*)d_out;          // also used as x2 residual storage

  // ---- workspace carve (~312 MB total) ----
  uint8_t* w = (uint8_t*)d_ws;
  auto carve = [&](size_t bytes) -> void* {
    uint8_t* p = w; w += (bytes + 255) & ~(size_t)255; return (void*)p;
  };
  bf16*  yln  = (bf16*)carve((size_t)MROWS * DN * 2);   // LN1 out, later LN2 out
  bf16*  Qb   = (bf16*)carve((size_t)MROWS * DN * 2);
  bf16*  Kb   = (bf16*)carve((size_t)MROWS * DN * 2);
  bf16*  Vb   = (bf16*)carve((size_t)MROWS * DN * 2);
  bf16*  Wqb  = (bf16*)carve((size_t)DN * DN * 2);
  bf16*  Wkb  = (bf16*)carve((size_t)DN * DN * 2);
  bf16*  Wvb  = (bf16*)carve((size_t)DN * DN * 2);
  bf16*  W1b  = (bf16*)carve((size_t)HIDN * DN * 2);
  bf16*  W2b  = (bf16*)carve((size_t)DN * HIDN * 2);
  float* corr = (float*)carve((size_t)BN * HN * LN * 4);
  float* o2d  = (float*)carve((size_t)BN * DN * 4);
  float* prj  = (float*)carve((size_t)BN * DN * 4);
  bf16*  h1   = (bf16*)carve((size_t)4096 * HIDN * 2);  // MLP hidden chunk (32 MB)

  // ---- weight conversion to bf16 ----
  cvt_kernel<<<(DN * DN) / 256, 256, 0, stream>>>(Wq, Wqb, DN * DN);
  cvt_kernel<<<(DN * DN) / 256, 256, 0, stream>>>(Wk, Wkb, DN * DN);
  cvt_kernel<<<(DN * DN) / 256, 256, 0, stream>>>(Wv, Wvb, DN * DN);
  cvt_kernel<<<(HIDN * DN) / 256, 256, 0, stream>>>(W1, W1b, HIDN * DN);
  cvt_kernel<<<(DN * HIDN) / 256, 256, 0, stream>>>(W2, W2b, DN * HIDN);

  // ---- LN1 ----
  ln_kernel<<<MROWS, 256, 0, stream>>>(x, ln1w, ln1b, yln);

  // ---- QKV projections (WMMA GEMM, bf16 out) ----
  dim3 gqkv(DN / 128, MROWS / 128);
  gemm_bf16<0><<<gqkv, 256, 0, stream>>>(yln, Wqb, bq, nullptr, Qb, MROWS, DN, DN);
  gemm_bf16<0><<<gqkv, 256, 0, stream>>>(yln, Wkb, bk, nullptr, Kb, MROWS, DN, DN);
  gemm_bf16<0><<<gqkv, 256, 0, stream>>>(yln, Wvb, bv, nullptr, Vb, MROWS, DN, DN);

  // ---- autocorrelation = tiled QK^T with wrapped-diagonal accumulation ----
  hipMemsetAsync(corr, 0, (size_t)BN * HN * LN * 4, stream);
  autocorr_kernel<<<dim3(LN / 128, LN / 128, BN * HN), 256, 0, stream>>>(Qb, Kb, corr);

  // ---- softmax(corr/HD), attn*V, output projection ----
  softmax_kernel<<<BN * HN, 256, 0, stream>>>(corr);
  attnv_kernel<<<BN * HN, 256, 0, stream>>>(corr, Vb, o2d);
  proj_kernel<<<(BN * DN) / 256, 256, 0, stream>>>(o2d, Wo, bo, prj);

  // ---- residual broadcast + series decomposition -> x2 (stored in d_out) ----
  decomp_kernel<<<(size_t)MROWS * DN / 256, 256, 0, stream>>>(x, prj, out);

  // ---- LN2 on x2 ----
  ln_kernel<<<MROWS, 256, 0, stream>>>(out, ln2w, ln2b, yln);

  // ---- MLP, chunked over M (8 chunks of 4096 rows) ----
  const int CH = 4096;
  for (int c = 0; c < MROWS / CH; ++c) {
    const bf16* a1 = yln + (size_t)c * CH * DN;
    float* resc = out + (size_t)c * CH * DN;
    gemm_bf16<2><<<dim3(HIDN / 128, CH / 128), 256, 0, stream>>>(
        a1, W1b, b1, nullptr, h1, CH, HIDN, DN);
    gemm_bf16<3><<<dim3(DN / 128, CH / 128), 256, 0, stream>>>(
        h1, W2b, b2, resc, resc, CH, DN, HIDN);
  }
}
